// HumanSender_27281632264216
// MI455X (gfx1250) — matrix operand order repaired and checked by reading, not compiled, not run on hardware
//
#include <hip/hip_runtime.h>
#include <hip/hip_bf16.h>

#define NN 50000
#define EE 800000
#define BSZ 1024
#define FIN 128
#define HID 128
#define EMBD 64
#define NREL 3
#define NBAS 2

typedef __bf16 bf16;
typedef __attribute__((ext_vector_type(16))) bf16  v16bf;
typedef __attribute__((ext_vector_type(8)))  float v8f;

// ---------------- weight prep: W[r] = sum_b comp[r][b]*basis[b]; stored transposed (OD x 128) bf16 ----------------
__global__ void prep_w(const float* __restrict__ basis, const float* __restrict__ comp,
                       const float* __restrict__ root, bf16* __restrict__ Wt, int OD) {
  int idx = blockIdx.x * blockDim.x + threadIdx.x;
  int per = OD * FIN;               // ID is always 128
  if (idx >= 4 * per) return;
  int mat = idx / per;
  int rem = idx - mat * per;
  int o = rem / FIN;
  int k = rem - o * FIN;
  float v;
  if (mat == 3) v = root[k * OD + o];
  else v = comp[mat * NBAS + 0] * basis[0 * FIN * OD + k * OD + o] +
           comp[mat * NBAS + 1] * basis[1 * FIN * OD + k * OD + o];
  Wt[idx] = (bf16)v;
}

__global__ void prep_fc(const float* __restrict__ fcw, bf16* __restrict__ Wt) {
  int idx = blockIdx.x * blockDim.x + threadIdx.x;
  if (idx >= 128 * 128) return;
  int o = idx >> 7, k = idx & 127;
  Wt[idx] = (bf16)fcw[k * 128 + o];
}

// ---------------- edge degree counts (shared by both layers), then reciprocal in place ----------------
__global__ void zero_f(float* __restrict__ p, int n) {
  int i = blockIdx.x * blockDim.x + threadIdx.x;
  if (i < n) p[i] = 0.0f;
}
__global__ void count_edges(const int* __restrict__ etype, const int* __restrict__ dst,
                            float* __restrict__ cnt) {
  int e = blockIdx.x * blockDim.x + threadIdx.x;
  if (e >= EE) return;
  atomicAdd(&cnt[etype[e] * NN + dst[e]], 1.0f);
}
__global__ void inv_cnt(float* __restrict__ cnt, int n) {
  int i = blockIdx.x * blockDim.x + threadIdx.x;
  if (i < n) cnt[i] = 1.0f / fmaxf(cnt[i], 1.0f);
}

// ---------------- WMMA fragment loads (ISA §7.12.2 layouts) ----------------
__device__ __forceinline__ v16bf load_a_frag(const float* __restrict__ A, int row, int kb, int lane) {
  int khalf = lane >> 4;                       // lanes 16-31 hold K = +8 / +24
  const float* p = A + (long)row * 128 + kb + khalf * 8;
  float4 a0 = *(const float4*)(p);
  float4 a1 = *(const float4*)(p + 4);
  float4 a2 = *(const float4*)(p + 16);
  float4 a3 = *(const float4*)(p + 20);
  v16bf a;
  a[0]=(bf16)a0.x; a[1]=(bf16)a0.y; a[2]=(bf16)a0.z; a[3]=(bf16)a0.w;
  a[4]=(bf16)a1.x; a[5]=(bf16)a1.y; a[6]=(bf16)a1.z; a[7]=(bf16)a1.w;
  a[8]=(bf16)a2.x; a[9]=(bf16)a2.y; a[10]=(bf16)a2.z; a[11]=(bf16)a2.w;
  a[12]=(bf16)a3.x; a[13]=(bf16)a3.y; a[14]=(bf16)a3.z; a[15]=(bf16)a3.w;
  return a;
}
__device__ __forceinline__ v16bf load_b_frag(const bf16* __restrict__ Bt, int col, int kb, int lane) {
  int khalf = lane >> 4;                       // lanes 16-31 hold K = +8 / +24
  const bf16* p = Bt + (long)col * 128 + kb + khalf * 8;
  union { uint4 u[2]; v16bf v; } bu;
  bu.u[0] = *(const uint4*)(p);
  bu.u[1] = *(const uint4*)(p + 16);
  return bu.v;
}

// ---------------- fused RGCN-layer GEMM: one A-fragment feeds 4 weight matrices ----------------
// Bt = [4][OD][128] bf16 (mats 0..2 = relations -> xr, mat 3 = root -> outRoot + bias)
template <int OD>
__global__ __launch_bounds__(256) void wmma_gemm_rgcn(const float* __restrict__ A,
                                                      const bf16* __restrict__ Bt,
                                                      const float* __restrict__ bias,
                                                      float* __restrict__ outRoot,
                                                      float* __restrict__ xr, int nrows) {
  const int lane   = threadIdx.x & 31;
  const int waveid = threadIdx.x >> 5;
  const int colTiles = OD / 16;
  const int tiles = (nrows / 16) * colTiles;
  int w = blockIdx.x * 8 + waveid;
  if (w >= tiles) return;                      // wave-uniform: EXEC stays all-1s for WMMA
  int rowTile = w / colTiles;
  int colTile = w - rowTile * colTiles;
  int row = rowTile * 16 + (lane & 15);
  int col = colTile * 16 + (lane & 15);

  v8f acc0 = {}, acc1 = {}, acc2 = {}, acc3 = {};
#pragma unroll
  for (int kb = 0; kb < 128; kb += 32) {
    v16bf a = load_a_frag(A, row, kb, lane);   // loaded/converted ONCE per kb
    v16bf b0 = load_b_frag(Bt + 0 * OD * 128, col, kb, lane);
    v16bf b1 = load_b_frag(Bt + 1 * OD * 128, col, kb, lane);
    v16bf b2 = load_b_frag(Bt + 2 * OD * 128, col, kb, lane);
    v16bf b3 = load_b_frag(Bt + 3 * OD * 128, col, kb, lane);
    acc0 = __builtin_amdgcn_wmma_f32_16x16x32_bf16(false, a, false, b0, (short)0, acc0, false, false);
    acc1 = __builtin_amdgcn_wmma_f32_16x16x32_bf16(false, a, false, b1, (short)0, acc1, false, false);
    acc2 = __builtin_amdgcn_wmma_f32_16x16x32_bf16(false, a, false, b2, (short)0, acc2, false, false);
    acc3 = __builtin_amdgcn_wmma_f32_16x16x32_bf16(false, a, false, b3, (short)0, acc3, false, false);
  }
  float bv = bias[col];
  int rbase = rowTile * 16 + ((lane >> 4) << 3);   // M = vgpr + 8*(lane>=16)
#pragma unroll
  for (int v = 0; v < 8; ++v) {
    long ro = (long)(rbase + v) * OD + col;
    outRoot[ro] = acc3[v] + bv;
    xr[(long)0 * nrows * OD + ro] = acc0[v];
    xr[(long)1 * nrows * OD + ro] = acc1[v];
    xr[(long)2 * nrows * OD + ro] = acc2[v];
  }
}

// ---------------- single-matrix WMMA GEMM (head): C = tanh(A*Bt^T + bias) ----------------
__global__ __launch_bounds__(256) void wmma_gemm_head(const float* __restrict__ A,
                                                      const bf16* __restrict__ Bt,
                                                      const float* __restrict__ bias,
                                                      float* __restrict__ C, int nrows) {
  const int lane   = threadIdx.x & 31;
  const int waveid = threadIdx.x >> 5;
  const int colTiles = 128 / 16;
  const int tiles = (nrows / 16) * colTiles;
  int w = blockIdx.x * 8 + waveid;
  if (w >= tiles) return;
  int rowTile = w / colTiles;
  int colTile = w - rowTile * colTiles;
  int row = rowTile * 16 + (lane & 15);
  int col = colTile * 16 + (lane & 15);

  v8f c = {};
#pragma unroll
  for (int kb = 0; kb < 128; kb += 32) {
    v16bf a = load_a_frag(A, row, kb, lane);
    v16bf b = load_b_frag(Bt, col, kb, lane);
    c = __builtin_amdgcn_wmma_f32_16x16x32_bf16(false, a, false, b, (short)0, c, false, false);
  }
  float bv = bias[col];
  int rbase = rowTile * 16 + ((lane >> 4) << 3);
#pragma unroll
  for (int v = 0; v < 8; ++v)
    C[(long)(rbase + v) * 128 + col] = tanhf(c[v] + bv);
}

// ---------------- edge scatter with normalization folded in: out[dst] += xr[r][src] * inv[r][dst] ----------------
template <int OD>
__global__ void scatter_edges(const int* __restrict__ etype, const int* __restrict__ src,
                              const int* __restrict__ dst, const float* __restrict__ xr,
                              const float* __restrict__ inv, float* __restrict__ out) {
  const int chunks = OD / 4;
  long t = (long)blockIdx.x * blockDim.x + threadIdx.x;
  long e = t / chunks;
  int  c = (int)(t - e * chunks) * 4;
  if (e >= EE) return;
  int r = etype[e], s = src[e], d = dst[e];
  float wgt = inv[(long)r * NN + d];
  float4 v = *(const float4*)(xr + ((long)r * NN + s) * OD + c);
  float* o = out + (long)d * OD + c;
  atomicAdd(o + 0, v.x * wgt);
  atomicAdd(o + 1, v.y * wgt);
  atomicAdd(o + 2, v.z * wgt);
  atomicAdd(o + 3, v.w * wgt);
}

__global__ void relu_k(float* __restrict__ p, int n) {
  int i = blockIdx.x * blockDim.x + threadIdx.x;
  if (i < n) p[i] = fmaxf(p[i], 0.0f);
}

__global__ void gather_pair(const float* __restrict__ emb, const int* __restrict__ nest,
                            const int* __restrict__ food, float* __restrict__ pair) {
  int t = blockIdx.x * blockDim.x + threadIdx.x;
  if (t >= BSZ * 128) return;
  int b = t >> 7, c = t & 127;
  int node = (c < EMBD) ? nest[b] : food[b];
  pair[t] = emb[(long)node * EMBD + (c & (EMBD - 1))];
}

extern "C" void kernel_launch(void* const* d_in, const int* in_sizes, int n_in,
                              void* d_out, int out_size, void* d_ws, size_t ws_size,
                              hipStream_t stream) {
  const float* x      = (const float*)d_in[0];
  const int*   ei     = (const int*)d_in[1];
  const int*   etype  = (const int*)d_in[3];
  const int*   nest   = (const int*)d_in[4];
  const int*   food   = (const int*)d_in[5];
  const float* basis1 = (const float*)d_in[6];
  const float* comp1  = (const float*)d_in[7];
  const float* root1  = (const float*)d_in[8];
  const float* bias1  = (const float*)d_in[9];
  const float* basis2 = (const float*)d_in[10];
  const float* comp2  = (const float*)d_in[11];
  const float* root2  = (const float*)d_in[12];
  const float* bias2  = (const float*)d_in[13];
  const float* fc_w   = (const float*)d_in[14];
  const float* fc_b   = (const float*)d_in[15];
  const int* src = ei;
  const int* dst = ei + EE;

  // workspace layout (256B aligned)
  char* ws = (char*)d_ws;
  size_t off = 0;
  auto alloc = [&](size_t bytes) -> void* {
    void* p = ws + off;
    off += (bytes + 255) & ~(size_t)255;
    return p;
  };
  bf16*  wt1   = (bf16*)alloc((size_t)4 * 128 * 128 * sizeof(bf16));   // [4][128][128]
  bf16*  wt2   = (bf16*)alloc((size_t)4 * 64 * 128 * sizeof(bf16));    // [4][64][128]
  bf16*  wtfc  = (bf16*)alloc((size_t)128 * 128 * sizeof(bf16));
  float* inv   = (float*)alloc((size_t)NREL * NN * sizeof(float));
  float* h     = (float*)alloc((size_t)NN * HID * sizeof(float));
  float* emb   = (float*)alloc((size_t)NN * EMBD * sizeof(float));
  float* xr    = (float*)alloc((size_t)NREL * NN * HID * sizeof(float)); // reused (layer2 uses NREL*NN*64)
  float* pairf = (float*)alloc((size_t)BSZ * 128 * sizeof(float));
  (void)ws_size; (void)in_sizes; (void)n_in; (void)out_size;

  const int T = 256;

  // 1) weights -> combined, transposed bf16
  prep_w<<<(4 * 128 * 128 + T - 1) / T, T, 0, stream>>>(basis1, comp1, root1, wt1, 128);
  prep_w<<<(4 * 64 * 128 + T - 1) / T, T, 0, stream>>>(basis2, comp2, root2, wt2, 64);
  prep_fc<<<(128 * 128 + T - 1) / T, T, 0, stream>>>(fc_w, wtfc);

  // 2) per-(relation,node) reciprocal in-degree (shared by both layers)
  zero_f<<<(NREL * NN + T - 1) / T, T, 0, stream>>>(inv, NREL * NN);
  count_edges<<<(EE + T - 1) / T, T, 0, stream>>>(etype, dst, inv);
  inv_cnt<<<(NREL * NN + T - 1) / T, T, 0, stream>>>(inv, NREL * NN);

  // 3) layer 1: fused 4-matrix GEMM (root+bias -> h, relations -> xr), normalized scatter, relu
  {
    int blocks = ((NN / 16) * (128 / 16) + 7) / 8;                 // 3125
    wmma_gemm_rgcn<128><<<blocks, T, 0, stream>>>(x, wt1, bias1, h, xr, NN);
    long st = (long)EE * (128 / 4);
    scatter_edges<128><<<(unsigned)((st + T - 1) / T), T, 0, stream>>>(etype, src, dst, xr, inv, h);
    relu_k<<<(NN * HID + T - 1) / T, T, 0, stream>>>(h, NN * HID);
  }

  // 4) layer 2: 128 -> 64
  {
    int blocks = ((NN / 16) * (64 / 16) + 7) / 8;                  // 1563
    wmma_gemm_rgcn<64><<<blocks, T, 0, stream>>>(h, wt2, bias2, emb, xr, NN);
    long st = (long)EE * (64 / 4);
    scatter_edges<64><<<(unsigned)((st + T - 1) / T), T, 0, stream>>>(etype, src, dst, xr, inv, emb);
  }

  // 5) head: gather pairs, GEMM + tanh -> d_out (1024 x 128)
  gather_pair<<<(BSZ * 128 + T - 1) / T, T, 0, stream>>>(emb, nest, food, pairf);
  {
    int blocks = ((BSZ / 16) * (128 / 16) + 7) / 8;                // 64
    wmma_gemm_head<<<blocks, T, 0, stream>>>(pairf, wtfc, fc_b, (float*)d_out, BSZ);
  }
}
